// HandPointnet_90202903150784
// MI455X (gfx1250) — matrix operand order repaired and checked by reading, not compiled, not run on hardware
//
#include <hip/hip_runtime.h>
#include <hip/hip_bf16.h>
#include <cstdint>

// ---------------------------------------------------------------------------
// PointNet++ (HandPointnet) forward for MI455X / gfx1250 (wave32, WMMA).
// Per SA module: FPS -> ball query -> fused(gather + MLP0 + MLP1 + maxpool).
// GEMMs: v_wmma_f32_16x16x32_f16, f16 activations (NHWC, cin padded to 32,
// channel order = [feats, rel_xyz, pad] so gathers are 16B-aligned), f16
// weights pre-packed with matching column permutation. BN+ReLU fused in the
// D-tile epilogue; max-over-K via wave32 shfl butterfly on accumulators.
// Gather feats + classifier activations use global_load_async_to_lds_b128.
// ---------------------------------------------------------------------------

typedef __attribute__((ext_vector_type(16))) _Float16 v16h;
typedef __attribute__((ext_vector_type(8)))  _Float16 v8h;
typedef __attribute__((ext_vector_type(8)))  float    v8f;

__device__ __forceinline__ unsigned ballot32(bool p) {
#if __has_builtin(__builtin_amdgcn_ballot_w32)
  return __builtin_amdgcn_ballot_w32(p);
#else
  return (unsigned)__ballot(p);
#endif
}

// ---- CDNA5 async global->LDS copy (ASYNCcnt-tracked) ----------------------
__device__ __forceinline__ void async_copy_b128(void* lds_dst, const void* gsrc) {
  unsigned lds = (unsigned)(uintptr_t)lds_dst;  // generic LDS addr: low 32 bits
  asm volatile("global_load_async_to_lds_b128 %0, %1, off"
               :: "v"(lds), "v"(gsrc) : "memory");
}
__device__ __forceinline__ void async_wait0() {
  asm volatile("s_wait_asynccnt 0x0" ::: "memory");
}

// ---- WMMA fragment helpers (16x16x32 f16, wave32) -------------------------
// A (16x32 MxK): lane l -> row m=l&15, half h=l>>4.
//   frag[0..7]  = W[m][h*8 + j]       frag[8..15] = W[m][16 + h*8 + j]
__device__ __forceinline__ v16h frag_a(const _Float16* Wtile, int stride, int lane) {
  const int m = lane & 15, h = lane >> 4;
  const _Float16* p = Wtile + m * stride + h * 8;
  v8h lo = *reinterpret_cast<const v8h*>(p);
  v8h hi = *reinterpret_cast<const v8h*>(p + 16);
  v16h f;
#pragma unroll
  for (int j = 0; j < 8; ++j) { f[j] = lo[j]; f[j + 8] = hi[j]; }
  return f;
}
// B (32x16 KxN): lane l -> col n=l&15, h=l>>4: frag[0..7]=X[n][h*8+j], frag[8..15]=X[n][16+h*8+j].
__device__ __forceinline__ v16h frag_b(const _Float16* Xtile, int stride, int lane) {
  const int n = lane & 15, h = lane >> 4;
  const _Float16* p = Xtile + n * stride + h * 8;
  v8h lo = *reinterpret_cast<const v8h*>(p);
  v8h hi = *reinterpret_cast<const v8h*>(p + 16);
  v16h f;
#pragma unroll
  for (int j = 0; j < 8; ++j) { f[j] = lo[j]; f[j + 8] = hi[j]; }
  return f;
}
__device__ __forceinline__ v8f wmma32(v16h a, v16h b, v8f c) {
  return __builtin_amdgcn_wmma_f32_16x16x32_f16(false, a, false, b, (short)0, c,
                                                false, false);
}

// ---------------------------------------------------------------------------
// prep: pointcloud [B][4096][9] -> xyz0 f32 [B][4096][3], feats0 f16 [B][4096][6]
__global__ __launch_bounds__(256) void prep_kernel(const float* __restrict__ pc,
                                                   float* __restrict__ xyz0,
                                                   _Float16* __restrict__ f0) {
  int i = blockIdx.x * 256 + threadIdx.x;
  if (i < 16 * 4096) {
    const float* p = pc + (size_t)i * 9;
    xyz0[i * 3 + 0] = p[0];
    xyz0[i * 3 + 1] = p[1];
    xyz0[i * 3 + 2] = p[2];
#pragma unroll
    for (int c = 0; c < 6; ++c) f0[i * 6 + c] = (_Float16)p[3 + c];
  }
}

// weights f32 [cout][cin] -> f16 [cout_pad][cin_pad], zero padded.
// perm>0: input-channel permutation [feats(perm), xyz(3)] (cin == perm+3).
__global__ __launch_bounds__(256) void convw_kernel(const float* __restrict__ W,
                                                    _Float16* __restrict__ Wh,
                                                    int cout, int cin,
                                                    int cout_pad, int cin_pad,
                                                    int perm) {
  int i = blockIdx.x * 256 + threadIdx.x;
  if (i < cout_pad * cin_pad) {
    int o = i / cin_pad, c = i % cin_pad;
    int c_old;
    if (perm > 0)
      c_old = (c < perm) ? (c + 3) : ((c < perm + 3) ? (c - perm) : -1);
    else
      c_old = (c < cin) ? c : -1;
    float v = (o < cout && c_old >= 0) ? W[o * cin + c_old] : 0.0f;
    Wh[i] = (_Float16)v;
  }
}

// ---------------------------------------------------------------------------
// Furthest point sampling: one workgroup/batch; wave32 shfl butterfly argmax,
// only 2 barriers per iteration. First-index tie-break matches jnp.argmax.
__global__ __launch_bounds__(256) void fps_kernel(const float* __restrict__ xyz,
                                                  int N, int S,
                                                  int* __restrict__ sidx,
                                                  float* __restrict__ new_xyz) {
  __shared__ float dist[4096];
  __shared__ float wrd[8];
  __shared__ int   wri[8];
  __shared__ int   s_far;
  const int b = blockIdx.x, tid = threadIdx.x;
  const int lane = tid & 31, wid = tid >> 5;
  const float* X = xyz + (size_t)b * N * 3;
  for (int p = tid; p < N; p += 256) dist[p] = 1e10f;
  if (tid == 0) s_far = 0;
  __syncthreads();
  for (int it = 0; it < S; ++it) {
    const int far = s_far;
    if (tid == 0) {
      sidx[b * S + it] = far;
      new_xyz[(b * S + it) * 3 + 0] = X[far * 3 + 0];
      new_xyz[(b * S + it) * 3 + 1] = X[far * 3 + 1];
      new_xyz[(b * S + it) * 3 + 2] = X[far * 3 + 2];
    }
    const float cx = X[far * 3 + 0], cy = X[far * 3 + 1], cz = X[far * 3 + 2];
    float bd = -1.0f; int bi = 0;
    for (int p = tid; p < N; p += 256) {
      float dx = X[p * 3 + 0] - cx, dy = X[p * 3 + 1] - cy, dz = X[p * 3 + 2] - cz;
      float d = dx * dx + dy * dy + dz * dz;
      float nd = fminf(dist[p], d);
      dist[p] = nd;
      if (nd > bd) { bd = nd; bi = p; }
    }
    // wave32 butterfly: max dist, tie -> smaller index
#pragma unroll
    for (int d = 16; d > 0; d >>= 1) {
      float od = __shfl_xor(bd, d, 32);
      int   oi = __shfl_xor(bi, d, 32);
      if (od > bd || (od == bd && oi < bi)) { bd = od; bi = oi; }
    }
    if (lane == 0) { wrd[wid] = bd; wri[wid] = bi; }
    __syncthreads();
    if (tid == 0) {
      float fb = wrd[0]; int fi = wri[0];
#pragma unroll
      for (int w2 = 1; w2 < 8; ++w2)
        if (wrd[w2] > fb || (wrd[w2] == fb && wri[w2] < fi)) { fb = wrd[w2]; fi = wri[w2]; }
      s_far = fi;
    }
    __syncthreads();
  }
}

// ---------------------------------------------------------------------------
// Ball query: one wave32 per center; ordered first-K-within-radius via
// ballot + prefix popcount; pad with first hit (0 if none).
__global__ __launch_bounds__(256) void ballquery_kernel(
    const float* __restrict__ xyz, const float* __restrict__ new_xyz,
    int* __restrict__ idx, int N, int S, int K, float r2, int ncenters) {
  const int wg = blockIdx.x * 8 + (threadIdx.x >> 5);
  if (wg >= ncenters) return;
  const int lane = threadIdx.x & 31;
  const int b = wg / S, s = wg % S;
  const float cx = new_xyz[(b * S + s) * 3 + 0];
  const float cy = new_xyz[(b * S + s) * 3 + 1];
  const float cz = new_xyz[(b * S + s) * 3 + 2];
  const float* X = xyz + (size_t)b * N * 3;
  int* out = idx + (size_t)(b * S + s) * K;
  int cnt = 0, first = -1;
  for (int base = 0; base < N; base += 32) {
    const int p = base + lane;
    bool pred = false;
    if (p < N) {
      float dx = X[p * 3 + 0] - cx, dy = X[p * 3 + 1] - cy, dz = X[p * 3 + 2] - cz;
      pred = (dx * dx + dy * dy + dz * dz) < r2;
    }
    unsigned mask = ballot32(pred);
    int prefix = __popc(mask & ((1u << lane) - 1u));
    if (pred && (cnt + prefix) < K) out[cnt + prefix] = p;
    if (first < 0 && mask) first = base + (__ffs(mask) - 1);
    cnt += __popc(mask);
    if (cnt >= K) break;
  }
  if (cnt < K) {
    int fill = (cnt == 0) ? 0 : first;
    for (int k = cnt + lane; k < K; k += 32) out[k] = fill;
  }
}

// ---------------------------------------------------------------------------
// Fused SA module. Block = 256 threads (8 waves); tile = TILE_S centers * K.
// Channel layout per column: [feats(CPREV), rel_xyz(3), zero pad] (weights
// are permuted to match), so feats copies are 16B aligned -> async B128.
template <int K, int TILE_S, int CPREV, int CIN_PAD, int C1, int C2, int NPREV>
__global__ __launch_bounds__(256) void sa_fused_kernel(
    const float* __restrict__ xyz_prev, const float* __restrict__ new_xyz,
    const _Float16* __restrict__ feats_prev, const int* __restrict__ idx,
    const _Float16* __restrict__ W0h, const float* __restrict__ g0,
    const float* __restrict__ b0, const _Float16* __restrict__ W1h,
    const float* __restrict__ g1, const float* __restrict__ b1,
    _Float16* __restrict__ feats_out, int S) {
  constexpr int TN = TILE_S * K;
  extern __shared__ char s_raw[];
  _Float16* xs = reinterpret_cast<_Float16*>(s_raw);  // [TN][CIN_PAD]
  _Float16* ys = xs + TN * CIN_PAD;                   // [TN][C1]
  const int sgroups = S / TILE_S;
  const int b = blockIdx.x / sgroups;
  const int s0 = (blockIdx.x % sgroups) * TILE_S;
  const int tid = threadIdx.x, lane = tid & 31, wid = tid >> 5;

  // ---- gather: feats (async B128 when aligned) + rel xyz + zero pad ----
  for (int n = tid; n < TN; n += 256) {
    const int sl = n / K, k = n % K;
    const int sg = s0 + sl;
    const int i = idx[(size_t)(b * S + sg) * K + k];
    const float* pp = xyz_prev + (size_t)(b * NPREV + i) * 3;
    const float* cc = new_xyz + (size_t)(b * S + sg) * 3;
    _Float16* col = xs + n * CIN_PAD;
    const _Float16* fp = feats_prev + (size_t)(b * NPREV + i) * CPREV;
    if constexpr (CPREV % 8 == 0) {
#pragma unroll
      for (int j = 0; j < CPREV / 8; ++j) async_copy_b128(col + j * 8, fp + j * 8);
    } else {
#pragma unroll
      for (int c = 0; c < CPREV; ++c) col[c] = fp[c];
    }
    col[CPREV + 0] = (_Float16)(pp[0] - cc[0]);
    col[CPREV + 1] = (_Float16)(pp[1] - cc[1]);
    col[CPREV + 2] = (_Float16)(pp[2] - cc[2]);
#pragma unroll
    for (int c = CPREV + 3; c < CIN_PAD; ++c) col[c] = (_Float16)0.0f;
  }
  if constexpr (CPREV % 8 == 0) async_wait0();
  __syncthreads();

  // ---- GEMM1: ys[n][o] = relu(W0*xs * g0 + b0); 2 N-tiles share one A-frag
  constexpr int NT1 = TN / 16, MT1 = C1 / 16, KT1 = CIN_PAD / 32;
  constexpr int NP = NT1 / 2;
  for (int t = wid; t < MT1 * NP; t += 8) {
    const int mt = t / NP, np = t % NP;
    v8f acc0 = {}, acc1 = {};
#pragma unroll
    for (int kt = 0; kt < KT1; ++kt) {
      v16h a = frag_a(W0h + mt * 16 * CIN_PAD + kt * 32, CIN_PAD, lane);
      v16h bb0 = frag_b(xs + (np * 32) * CIN_PAD + kt * 32, CIN_PAD, lane);
      v16h bb1 = frag_b(xs + (np * 32 + 16) * CIN_PAD + kt * 32, CIN_PAD, lane);
      acc0 = wmma32(a, bb0, acc0);
      acc1 = wmma32(a, bb1, acc1);
    }
    const int n = lane & 15, mset = (lane >> 4) * 8;
    v8h o0, o1;
#pragma unroll
    for (int r = 0; r < 8; ++r) {
      const int o = mt * 16 + mset + r;
      const float gv = g0[o], bv = b0[o];
      float v0 = acc0[r] * gv + bv, v1 = acc1[r] * gv + bv;
      o0[r] = (_Float16)(v0 > 0.0f ? v0 : 0.0f);
      o1[r] = (_Float16)(v1 > 0.0f ? v1 : 0.0f);
    }
    *reinterpret_cast<v8h*>(ys + (np * 32 + n) * C1 + mt * 16 + mset) = o0;
    *reinterpret_cast<v8h*>(ys + (np * 32 + 16 + n) * C1 + mt * 16 + mset) = o1;
  }
  __syncthreads();

  // ---- GEMM2 + BN/ReLU + max over K; A-frag reused across all KSUB tiles
  constexpr int KT2 = C1 / 32, KSUB = K / 16;
  constexpr int UNITS = (C2 / 16) * TILE_S;
  for (int u = wid; u < UNITS; u += 8) {
    const int mt = u / TILE_S, sl = u % TILE_S;
    v8f acc[KSUB] = {};
    for (int kt = 0; kt < KT2; ++kt) {
      v16h a = frag_a(W1h + mt * 16 * C1 + kt * 32, C1, lane);
#pragma unroll
      for (int ks = 0; ks < KSUB; ++ks) {
        v16h bb = frag_b(ys + ((sl * KSUB + ks) * 16) * C1 + kt * 32, C1, lane);
        acc[ks] = wmma32(a, bb, acc[ks]);
      }
    }
    const int mset = (lane >> 4) * 8;
    float mx[8];
#pragma unroll
    for (int r = 0; r < 8; ++r) mx[r] = 0.0f;  // post-ReLU >= 0
#pragma unroll
    for (int r = 0; r < 8; ++r) {
      const int o = mt * 16 + mset + r;
      const float gv = g1[o], bv = b1[o];
#pragma unroll
      for (int ks = 0; ks < KSUB; ++ks) {
        float v = acc[ks][r] * gv + bv;
        v = v > 0.0f ? v : 0.0f;
        mx[r] = fmaxf(mx[r], v);
      }
    }
    // reduce across 16 columns held in each 16-lane half
#pragma unroll
    for (int d = 1; d < 16; d <<= 1)
#pragma unroll
      for (int r = 0; r < 8; ++r) mx[r] = fmaxf(mx[r], __shfl_xor(mx[r], d, 32));
    if ((lane & 15) == 0) {
      _Float16* fo = feats_out + (size_t)(b * S + s0 + sl) * C2 + mt * 16 + mset;
#pragma unroll
      for (int r = 0; r < 8; ++r) fo[r] = (_Float16)mx[r];
    }
  }
}

// ---------------------------------------------------------------------------
// Classifier head: activations [16 batches][<=1024 ch] double-buffered in LDS.
__device__ __forceinline__ void cls_layer(const _Float16* Wh, const float* g,
                                          const float* bia, const _Float16* src,
                                          _Float16* dst, int cinp, int cout,
                                          int lane, int wid) {
  for (int t = wid; t < cout / 16; t += 8) {
    v8f acc = {};
    for (int kt = 0; kt < cinp / 32; ++kt) {
      v16h a = frag_a(Wh + t * 16 * cinp + kt * 32, cinp, lane);
      v16h bb = frag_b(src + kt * 32, 1024, lane);
      acc = wmma32(a, bb, acc);
    }
    const int n = lane & 15, mset = (lane >> 4) * 8;
    v8h o8;
#pragma unroll
    for (int r = 0; r < 8; ++r) {
      const int o = t * 16 + mset + r;
      float v = acc[r] * g[o] + bia[o];
      o8[r] = (_Float16)(v > 0.0f ? v : 0.0f);
    }
    *reinterpret_cast<v8h*>(dst + n * 1024 + t * 16 + mset) = o8;
  }
}

__global__ __launch_bounds__(256) void cls_kernel(
    const _Float16* __restrict__ feats,  // [16][1024] f16
    const _Float16* W0, const float* g0, const float* b0,
    const _Float16* W1, const float* g1, const float* b1,
    const _Float16* W2, const float* g2, const float* b2,
    const _Float16* W3, const float* b3, float* __restrict__ out) {
  extern __shared__ char s_raw[];
  _Float16* xb = reinterpret_cast<_Float16*>(s_raw);  // [16][1024]
  _Float16* yb = xb + 16 * 1024;
  const int tid = threadIdx.x, lane = tid & 31, wid = tid >> 5;
  for (int i = tid * 8; i < 16 * 1024; i += 256 * 8)
    async_copy_b128(xb + i, feats + i);
  async_wait0();
  __syncthreads();
  cls_layer(W0, g0, b0, xb, yb, 1024, 1024, lane, wid);
  __syncthreads();
  cls_layer(W1, g1, b1, yb, xb, 1024, 512, lane, wid);
  __syncthreads();
  cls_layer(W2, g2, b2, xb, yb, 512, 128, lane, wid);
  __syncthreads();
  // final: [64(63)][128], bias only, f32 out [16][63]
  for (int t = wid; t < 4; t += 8) {
    v8f acc = {};
    for (int kt = 0; kt < 4; ++kt) {
      v16h a = frag_a(W3 + t * 16 * 128 + kt * 32, 128, lane);
      v16h bb = frag_b(yb + kt * 32, 1024, lane);
      acc = wmma32(a, bb, acc);
    }
    const int n = lane & 15, mset = (lane >> 4) * 8;
#pragma unroll
    for (int r = 0; r < 8; ++r) {
      const int o = t * 16 + mset + r;
      if (o < 63) out[n * 63 + o] = acc[r] + b3[o];
    }
  }
}

// ---------------------------------------------------------------------------
extern "C" void kernel_launch(void* const* d_in, const int* in_sizes, int n_in,
                              void* d_out, int out_size, void* d_ws, size_t ws_size,
                              hipStream_t stream) {
  (void)in_sizes; (void)n_in; (void)out_size; (void)ws_size;
  const float* pc = (const float*)d_in[0];
  const float* Wf[10] = {(const float*)d_in[1],  (const float*)d_in[4],
                         (const float*)d_in[7],  (const float*)d_in[10],
                         (const float*)d_in[13], (const float*)d_in[16],
                         (const float*)d_in[19], (const float*)d_in[22],
                         (const float*)d_in[25], (const float*)d_in[28]};
  const float* Gf[9] = {(const float*)d_in[2],  (const float*)d_in[5],
                        (const float*)d_in[8],  (const float*)d_in[11],
                        (const float*)d_in[14], (const float*)d_in[17],
                        (const float*)d_in[20], (const float*)d_in[23],
                        (const float*)d_in[26]};
  const float* Bf[10] = {(const float*)d_in[3],  (const float*)d_in[6],
                         (const float*)d_in[9],  (const float*)d_in[12],
                         (const float*)d_in[15], (const float*)d_in[18],
                         (const float*)d_in[21], (const float*)d_in[24],
                         (const float*)d_in[27], (const float*)d_in[29]};
  float* out = (float*)d_out;

  static const int coutA[10] = {64, 128, 128, 256, 512, 1024, 1024, 512, 128, 63};
  static const int cinA[10]  = {9, 64, 131, 128, 259, 512, 1024, 1024, 512, 128};
  static const int coutP[10] = {64, 128, 128, 256, 512, 1024, 1024, 512, 128, 64};
  static const int cinP[10]  = {32, 64, 160, 128, 288, 512, 1024, 1024, 512, 128};
  static const int permA[10] = {6, 0, 128, 0, 256, 0, 0, 0, 0, 0};

  // ---- workspace carve-out (256B aligned) ----
  char* w = (char*)d_ws;
  auto alloc = [&](size_t bytes) -> void* {
    void* p = (void*)w;
    w += (bytes + 255) & ~(size_t)255;
    return p;
  };
  float*     xyz0   = (float*)alloc((size_t)16 * 4096 * 3 * 4);
  _Float16*  feats0 = (_Float16*)alloc((size_t)16 * 4096 * 6 * 2);
  _Float16*  Wh[10];
  for (int l = 0; l < 10; ++l)
    Wh[l] = (_Float16*)alloc((size_t)coutP[l] * cinP[l] * 2);
  int*       sidx   = (int*)alloc((size_t)16 * 512 * 4);
  float*     nxyz1  = (float*)alloc((size_t)16 * 512 * 3 * 4);
  float*     nxyz2  = (float*)alloc((size_t)16 * 64 * 3 * 4);
  float*     nxyz3  = (float*)alloc((size_t)16 * 1 * 3 * 4);
  int*       bidx   = (int*)alloc((size_t)16 * 512 * 32 * 4);  // reused all levels
  _Float16*  feats1 = (_Float16*)alloc((size_t)16 * 512 * 128 * 2);
  _Float16*  feats2 = (_Float16*)alloc((size_t)16 * 64 * 256 * 2);
  _Float16*  feats3 = (_Float16*)alloc((size_t)16 * 1024 * 2);

  // ---- prep & weight packing ----
  prep_kernel<<<(16 * 4096 + 255) / 256, 256, 0, stream>>>(pc, xyz0, feats0);
  for (int l = 0; l < 10; ++l) {
    int tot = coutP[l] * cinP[l];
    convw_kernel<<<(tot + 255) / 256, 256, 0, stream>>>(
        Wf[l], Wh[l], coutA[l], cinA[l], coutP[l], cinP[l], permA[l]);
  }

  // ---- SA level 0: N=4096 -> S=512, K=32, cin 9->64->128 ----
  fps_kernel<<<16, 256, 0, stream>>>(xyz0, 4096, 512, sidx, nxyz1);
  ballquery_kernel<<<(16 * 512 + 7) / 8, 256, 0, stream>>>(
      xyz0, nxyz1, bidx, 4096, 512, 32, 0.1f * 0.1f, 16 * 512);
  {
    const size_t sh = (size_t)(256 * 32 + 256 * 64) * 2;  // 48 KB
    auto kfn = sa_fused_kernel<32, 8, 6, 32, 64, 128, 4096>;
    hipFuncSetAttribute((const void*)kfn, hipFuncAttributeMaxDynamicSharedMemorySize,
                        (int)sh);
    kfn<<<16 * (512 / 8), 256, sh, stream>>>(xyz0, nxyz1, feats0, bidx, Wh[0],
                                             Gf[0], Bf[0], Wh[1], Gf[1], Bf[1],
                                             feats1, 512);
  }

  // ---- SA level 1: N=512 -> S=64, K=64, cin 131->128->256 ----
  fps_kernel<<<16, 256, 0, stream>>>(nxyz1, 512, 64, sidx, nxyz2);
  ballquery_kernel<<<(16 * 64 + 7) / 8, 256, 0, stream>>>(
      nxyz1, nxyz2, bidx, 512, 64, 64, 0.2f * 0.2f, 16 * 64);
  {
    const size_t sh = (size_t)(256 * 160 + 256 * 128) * 2;  // 144 KB
    auto kfn = sa_fused_kernel<64, 4, 128, 160, 128, 256, 512>;
    hipFuncSetAttribute((const void*)kfn, hipFuncAttributeMaxDynamicSharedMemorySize,
                        (int)sh);
    kfn<<<16 * (64 / 4), 256, sh, stream>>>(nxyz1, nxyz2, feats1, bidx, Wh[2],
                                            Gf[2], Bf[2], Wh[3], Gf[3], Bf[3],
                                            feats2, 64);
  }

  // ---- SA level 2: N=64 -> S=1, K=64, cin 259->512->1024 ----
  fps_kernel<<<16, 256, 0, stream>>>(nxyz2, 64, 1, sidx, nxyz3);
  ballquery_kernel<<<(16 * 1 + 7) / 8, 256, 0, stream>>>(
      nxyz2, nxyz3, bidx, 64, 1, 64, 0.4f * 0.4f, 16 * 1);
  {
    const size_t sh = (size_t)(64 * 288 + 64 * 512) * 2;  // 100 KB
    auto kfn = sa_fused_kernel<64, 1, 256, 288, 512, 1024, 64>;
    hipFuncSetAttribute((const void*)kfn, hipFuncAttributeMaxDynamicSharedMemorySize,
                        (int)sh);
    kfn<<<16, 256, sh, stream>>>(nxyz2, nxyz3, feats2, bidx, Wh[4], Gf[4], Bf[4],
                                 Wh[5], Gf[5], Bf[5], feats3, 1);
  }

  // ---- classifier head: [16][1024] -> [16][63] ----
  {
    const size_t sh = (size_t)2 * 16 * 1024 * 2;  // 64 KB
    hipFuncSetAttribute((const void*)cls_kernel,
                        hipFuncAttributeMaxDynamicSharedMemorySize, (int)sh);
    cls_kernel<<<1, 256, sh, stream>>>(feats3, Wh[6], Gf[6], Bf[6], Wh[7], Gf[7],
                                       Bf[7], Wh[8], Gf[8], Bf[8], Wh[9], Bf[9],
                                       out);
  }
}
